// HYLA_90744069030139
// MI455X (gfx1250) — matrix operand order repaired and checked by reading, not compiled, not run on hardware
//
#include <hip/hip_runtime.h>
#include <hip/hip_bf16.h>
#include <math.h>

// ---------------------------------------------------------------------------
// Types
// ---------------------------------------------------------------------------
typedef __attribute__((ext_vector_type(8)))  __bf16 v8bf;
typedef __attribute__((ext_vector_type(16))) __bf16 v16bf;
typedef __attribute__((ext_vector_type(8)))  float  v8f;

__device__ __forceinline__ __bf16 f2bf(float x) {
    unsigned u = __builtin_bit_cast(unsigned, x);
    unsigned r = (u + 0x7FFFu + ((u >> 16) & 1u)) >> 16;   // RNE
    return __builtin_bit_cast(__bf16, (unsigned short)r);
}
__device__ __forceinline__ v8f zero8() {
    v8f z;
#pragma unroll
    for (int i = 0; i < 8; ++i) z[i] = 0.0f;
    return z;
}
__device__ __forceinline__ v8bf zero8bf() {
    v8bf z;
#pragma unroll
    for (int i = 0; i < 8; ++i) z[i] = __builtin_bit_cast(__bf16, (unsigned short)0);
    return z;
}

// XOR-lane shuffle via ds_swizzle (group-of-32 mode: and=0x1F, or=0, xor=XM).
// Cheaper than the generic __shfl_xor lowering (single DS op, no addr setup).
template <int XM>
__device__ __forceinline__ float shx(float x) {
    return __builtin_bit_cast(float,
        __builtin_amdgcn_ds_swizzle(__builtin_bit_cast(int, x), (XM << 10) | 0x1F));
}
__device__ __forceinline__ float sum16(float p) {
    p += shx<1>(p); p += shx<2>(p); p += shx<4>(p); p += shx<8>(p);
    return p;
}
__device__ __forceinline__ float max16(float p) {
    p = fmaxf(p, shx<1>(p)); p = fmaxf(p, shx<2>(p));
    p = fmaxf(p, shx<4>(p)); p = fmaxf(p, shx<8>(p));
    return p;
}

// ---------------------------------------------------------------------------
// CDNA5 async global->LDS copy (ASYNCcnt-tracked, bypasses VGPRs).
// GVS mode: mem_addr = SADDR(base) + VADDR(byte offset). VDST = LDS byte addr.
// Generic pointers to LDS carry the wave-relative LDS offset in addr[31:0].
// ---------------------------------------------------------------------------
__device__ __forceinline__ unsigned lds_off(const void* p) {
    return (unsigned)(unsigned long long)p;
}
__device__ __forceinline__ void async_b128(const void* gbase, unsigned byte_off,
                                           unsigned lds) {
    asm volatile("global_load_async_to_lds_b128 %0, %1, %2"
                 :: "v"(lds), "v"(byte_off),
                    "s"((unsigned long long)(size_t)gbase)
                 : "memory");
}
__device__ __forceinline__ void wait_async() {
    asm volatile("s_wait_asynccnt 0x0" ::: "memory");
}

// A-fragment (16x32 bf16, M x K). ISA: lane<16 row=lane K={0..7,16..23};
// lane>=16 same row, K={8..15,24..31}. Two b128 LDS reads.
__device__ __forceinline__ v16bf load_fragA(const __bf16* base, int stride) {
    const int lane = threadIdx.x & 31;
    const int r = lane & 15, g = lane >> 4;
    const __bf16* p = base + r * stride + g * 8;
    v8bf lo = *reinterpret_cast<const v8bf*>(p);
    v8bf hi = *reinterpret_cast<const v8bf*>(p + 16);
    v16bf f;
#pragma unroll
    for (int i = 0; i < 8; ++i) { f[i] = lo[i]; f[i + 8] = hi[i]; }
    return f;
}

// B-fragment (32x16 bf16, K x N), operand stored rows = N-columns, K contiguous.
// ISA: lane<16 col=lane K={0..7,16..23}; lane>=16 col=lane-16 K={8..15,24..31}.
__device__ __forceinline__ v16bf load_fragB(const __bf16* base, int stride) {
    const int lane = threadIdx.x & 31;
    const int n = lane & 15, g = lane >> 4;
    const __bf16* p = base + n * stride + g * 8;
    v8bf lo = *reinterpret_cast<const v8bf*>(p);
    v8bf hi = *reinterpret_cast<const v8bf*>(p + 16);
    v16bf f;
#pragma unroll
    for (int i = 0; i < 8; ++i) { f[i] = lo[i]; f[i + 8] = hi[i]; }
    return f;
}

__device__ __forceinline__ v8f wmma_bf16(v16bf a, v16bf b, v8f c) {
    return __builtin_amdgcn_wmma_f32_16x16x32_bf16(
        false, a, false, b, (short)0, c, false, false);
}

// ---------------------------------------------------------------------------
// Prep kernels: weight transpose f32 -> [N][K] bf16, tanh(v_init)
// ---------------------------------------------------------------------------
__global__ __launch_bounds__(256) void k_transpose(
    const float* __restrict__ W, __bf16* __restrict__ WT, int K, int N) {
    __shared__ float t[32][33];
    const int bx = blockIdx.x * 32;  // N direction
    const int by = blockIdx.y * 32;  // K direction
    const int tx = threadIdx.x, ty = threadIdx.y;
#pragma unroll
    for (int i = 0; i < 32; i += 8)
        t[ty + i][tx] = W[(size_t)(by + ty + i) * N + bx + tx];
    __syncthreads();
#pragma unroll
    for (int i = 0; i < 32; i += 8)
        WT[(size_t)(bx + ty + i) * K + by + tx] = f2bf(t[tx][ty + i]);
}

__global__ void k_tanhv(const float* __restrict__ vinit, float* __restrict__ tv, int n) {
    int i = blockIdx.x * blockDim.x + threadIdx.x;
    if (i < n) tv[i] = tanhf(vinit[i]);
}

// ---------------------------------------------------------------------------
// WMMA GEMM: C[M,N] = A[M,512] * W (W transposed [N][512] bf16) + bias
// BM=128, BN=64, BK=32, 256 threads = 8 waves; wave w owns 16 rows x 64 cols
// (= one head group -> fused per-head L2 norm). Double-buffered LDS; tile k+1
// streams in via global_load_async_to_lds while tile k is consumed by WMMA.
// EPI: 0 norm->bf16 out0 (qh)
//      1 colBase<512: norm->out0 (kh); else bias->bf16 out1 (v_lin)
//      2 relu(sigmoid(acc+b)*tanh(v_init)) then norm -> out0 (v)
//      3 acc + b + shortcut(extra f32) -> f32 out0
// ---------------------------------------------------------------------------
template <int EPI, bool AF32>
__global__ __launch_bounds__(256) void k_gemm(
    const void* __restrict__ Aptr, const __bf16* __restrict__ BT,
    const float* __restrict__ bias, void* __restrict__ out0,
    void* __restrict__ out1, const float* __restrict__ extra) {
    __shared__ __align__(16) __bf16 As[2][128][40];
    __shared__ __align__(16) __bf16 Bs[2][64][40];

    const int tid = threadIdx.x;
    const int w = tid >> 5, lane = tid & 31;
    const int ln = lane & 15, lg = lane >> 4;
    const int rowBase = blockIdx.x * 128;
    const int colBase = blockIdx.y * 64;

    // staging coordinates
    const int ar = tid >> 1, ac0 = (tid & 1) * 16;  // A: 16 elems/thread
    const int br = tid >> 2, bc0 = (tid & 3) * 8;   // B: 8 elems/thread

    auto stage = [&](int kb, int p) {
        if (AF32) {  // convert f32 -> bf16 through VGPRs
            const float* A = (const float*)Aptr;
            const float* src = A + (size_t)(rowBase + ar) * 512 + kb + ac0;
#pragma unroll
            for (int i = 0; i < 4; ++i) {
                float4 vv = *reinterpret_cast<const float4*>(src + i * 4);
                As[p][ar][ac0 + i * 4 + 0] = f2bf(vv.x);
                As[p][ar][ac0 + i * 4 + 1] = f2bf(vv.y);
                As[p][ar][ac0 + i * 4 + 2] = f2bf(vv.z);
                As[p][ar][ac0 + i * 4 + 3] = f2bf(vv.w);
            }
        } else {     // bf16: async DMA straight into LDS
            unsigned off = ((unsigned)(rowBase + ar) * 512u + kb + ac0) * 2u;
            unsigned l0 = lds_off(&As[p][ar][ac0]);
            async_b128(Aptr, off, l0);
            async_b128(Aptr, off + 16u, l0 + 16u);
        }
        unsigned boff = ((unsigned)(colBase + br) * 512u + kb + bc0) * 2u;
        async_b128(BT, boff, lds_off(&Bs[p][br][bc0]));
    };

    v8f acc[4];
#pragma unroll
    for (int j = 0; j < 4; ++j) acc[j] = zero8();

    stage(0, 0);
    wait_async();
    __syncthreads();

    int p = 0;
    for (int kb = 0; kb < 512; kb += 32, p ^= 1) {
        if (kb + 32 < 512) stage(kb + 32, p ^ 1);   // prefetch next tile
        v16bf a = load_fragA(&As[p][w * 16][0], 40);
#pragma unroll
        for (int j = 0; j < 4; ++j) {
            v16bf b = load_fragB(&Bs[p][j * 16][0], 40);
            acc[j] = wmma_bf16(a, b, acc[j]);
        }
        wait_async();
        __syncthreads();
    }

    // ---------------- epilogue ----------------
    const int rowW = rowBase + w * 16;
    float bv[4];
#pragma unroll
    for (int j = 0; j < 4; ++j) bv[j] = bias[colBase + j * 16 + ln];

    if (EPI == 3) {
        float* O = (float*)out0;
        const float* S = extra;  // residual shortcut (f32)
#pragma unroll
        for (int j = 0; j < 4; ++j)
#pragma unroll
            for (int v = 0; v < 8; ++v) {
                const int row = rowW + v + 8 * lg;
                const int col = colBase + j * 16 + ln;
                const size_t idx = (size_t)row * 512 + col;
                O[idx] = acc[j][v] + bv[j] + S[idx];
            }
        return;
    }

    if (EPI == 1 && colBase >= 512) {  // v_lin passthrough (bf16)
        __bf16* O = (__bf16*)out1;
#pragma unroll
        for (int j = 0; j < 4; ++j)
#pragma unroll
            for (int v = 0; v < 8; ++v) {
                const int row = rowW + v + 8 * lg;
                const int col = (colBase - 512) + j * 16 + ln;
                O[(size_t)row * 512 + col] = f2bf(acc[j][v] + bv[j]);
            }
        return;
    }

    if (EPI == 2) {  // relu(sigmoid(acc+b) * tanh(v_init))
#pragma unroll
        for (int j = 0; j < 4; ++j)
#pragma unroll
            for (int v = 0; v < 8; ++v) {
                const int row = rowW + v + 8 * lg;
                const int col = colBase + j * 16 + ln;
                const float g = 1.0f / (1.0f + __expf(-(acc[j][v] + bv[j])));
                const float tv = extra[(size_t)(row % 24) * 512 + col];
                acc[j][v] = fmaxf(g * tv, 0.0f);
            }
    } else {
#pragma unroll
        for (int j = 0; j < 4; ++j)
#pragma unroll
            for (int v = 0; v < 8; ++v) acc[j][v] += bv[j];
    }

    // per-head (64-col) L2 norm: reduce sum-of-squares across the 16 lanes
    // holding the same row (xor masks 1..8 stay inside each half-wave).
    float rn[8];
#pragma unroll
    for (int v = 0; v < 8; ++v) {
        float s = 0.0f;
#pragma unroll
        for (int j = 0; j < 4; ++j) s += acc[j][v] * acc[j][v];
        s = sum16(s);
        rn[v] = 1.0f / fmaxf(sqrtf(s), 1e-12f);
    }
    __bf16* O = (__bf16*)out0;
#pragma unroll
    for (int j = 0; j < 4; ++j)
#pragma unroll
        for (int v = 0; v < 8; ++v) {
            const int row = rowW + v + 8 * lg;
            const int col = colBase + j * 16 + ln;
            O[(size_t)row * 512 + col] = f2bf(acc[j][v] * rn[v]);
        }
}

// ---------------------------------------------------------------------------
// Attention: one wave per (bt, h). l=24 padded to 32 with masking.
// S = q k^T / 8 -> masked softmax -> x = P v.  All matmuls via WMMA bf16.
// q/k tiles stream in with async global->LDS (wave-private: own asynccnt wait,
// no barrier needed; same-wave DS ops are pipeline-ordered).
// ---------------------------------------------------------------------------
__global__ __launch_bounds__(64) void k_attn(
    const __bf16* __restrict__ qh, const __bf16* __restrict__ kh,
    const __bf16* __restrict__ vv, __bf16* __restrict__ xs) {
    __shared__ __align__(16) __bf16 qs[2][32][72];
    __shared__ __align__(16) __bf16 ks[2][32][72];
    __shared__ __align__(16) __bf16 vts[2][64][40];  // v^T: rows=d, cols=l
    __shared__ __align__(16) __bf16 ps[2][32][40];

    const int w = threadIdx.x >> 5, lane = threadIdx.x & 31;
    const int ln = lane & 15, lg = lane >> 4;
    const int gw = blockIdx.x * 2 + w;   // 0..16383
    const int bt = gw >> 3, h = gw & 7;
    const size_t base = (size_t)bt * 24 * 512 + (size_t)h * 64;
    const unsigned base2 = (unsigned)base * 2u;  // byte offset (fits 32-bit)

    const v8bf z8 = zero8bf();

    // async-load q, k tiles (24 rows x 64 cols); zero-pad rows 24..31
    for (int i = lane; i < 24 * 8; i += 32) {
        const int r = i >> 3, ch = (i & 7) * 8;
        const unsigned off = base2 + (unsigned)(r * 512 + ch) * 2u;
        async_b128(qh, off, lds_off(&qs[w][r][ch]));
        async_b128(kh, off, lds_off(&ks[w][r][ch]));
    }
    for (int i = lane; i < 8 * 8; i += 32) {
        const int r = 24 + (i >> 3), ch = (i & 7) * 8;
        *reinterpret_cast<v8bf*>(&qs[w][r][ch]) = z8;
        *reinterpret_cast<v8bf*>(&ks[w][r][ch]) = z8;
    }
    // v loaded transposed (needs lane shuffle -> VGPR path); pad cols 24..31
    for (int i = lane; i < 24 * 8; i += 32) {
        const int r = i >> 3, ch = (i & 7) * 8;
        v8bf t = *reinterpret_cast<const v8bf*>(vv + base + (size_t)r * 512 + ch);
#pragma unroll
        for (int e = 0; e < 8; ++e) vts[w][ch + e][r] = t[e];
    }
    for (int i = lane; i < 64 * 8; i += 32)
        vts[w][i >> 3][24 + (i & 7)] = __builtin_bit_cast(__bf16, (unsigned short)0);
    wait_async();  // q/k tiles resident before fragment reads

    // S = q k^T (K = d = 64, two steps of 32)
    v8f s[2][2];
#pragma unroll
    for (int mi = 0; mi < 2; ++mi)
#pragma unroll
        for (int ni = 0; ni < 2; ++ni) s[mi][ni] = zero8();
#pragma unroll
    for (int kk = 0; kk < 64; kk += 32) {
        v16bf a0 = load_fragA(&qs[w][0][kk], 72);
        v16bf a1 = load_fragA(&qs[w][16][kk], 72);
        v16bf b0 = load_fragB(&ks[w][0][kk], 72);
        v16bf b1 = load_fragB(&ks[w][16][kk], 72);
        s[0][0] = wmma_bf16(a0, b0, s[0][0]);
        s[0][1] = wmma_bf16(a0, b1, s[0][1]);
        s[1][0] = wmma_bf16(a1, b0, s[1][0]);
        s[1][1] = wmma_bf16(a1, b1, s[1][1]);
    }

    // masked softmax over columns (cols 24..31 invalid -> -1e30)
    const float scale = 0.125f;  // 1/sqrt(64)
    const bool mcol = (ln >= 8);
#pragma unroll
    for (int mi = 0; mi < 2; ++mi)
#pragma unroll
        for (int v = 0; v < 8; ++v) {
            const float a = s[mi][0][v] * scale;
            const float b = mcol ? -1e30f : s[mi][1][v] * scale;
            const float mx = max16(fmaxf(a, b));
            const float ea = __expf(a - mx);
            const float eb = mcol ? 0.0f : __expf(b - mx);
            const float rr = 1.0f / sum16(ea + eb);
            const int row = mi * 16 + v + 8 * lg;
            ps[w][row][ln] = f2bf(ea * rr);
            ps[w][row][16 + ln] = f2bf(eb * rr);
        }

    // x = P v  (K = l padded to 32, one step)
    v8f x[2][4];
#pragma unroll
    for (int mi = 0; mi < 2; ++mi)
#pragma unroll
        for (int nj = 0; nj < 4; ++nj) x[mi][nj] = zero8();
    v16bf p0 = load_fragA(&ps[w][0][0], 40);
    v16bf p1 = load_fragA(&ps[w][16][0], 40);
#pragma unroll
    for (int nj = 0; nj < 4; ++nj) {
        v16bf b = load_fragB(&vts[w][nj * 16][0], 40);
        x[0][nj] = wmma_bf16(p0, b, x[0][nj]);
        x[1][nj] = wmma_bf16(p1, b, x[1][nj]);
    }
    // store rows 0..23
#pragma unroll
    for (int mi = 0; mi < 2; ++mi)
#pragma unroll
        for (int nj = 0; nj < 4; ++nj)
#pragma unroll
            for (int v = 0; v < 8; ++v) {
                const int row = mi * 16 + v + 8 * lg;
                if (row < 24)
                    xs[base + (size_t)row * 512 + nj * 16 + ln] = f2bf(x[mi][nj][v]);
            }
}

// ---------------------------------------------------------------------------
// Launch
// ---------------------------------------------------------------------------
extern "C" void kernel_launch(void* const* d_in, const int* in_sizes, int n_in,
                              void* d_out, int out_size, void* d_ws, size_t ws_size,
                              hipStream_t stream) {
    (void)in_sizes; (void)n_in; (void)out_size; (void)ws_size;
    const float* q     = (const float*)d_in[0];
    const float* kv    = (const float*)d_in[1];
    const float* Wq    = (const float*)d_in[2];
    const float* bq    = (const float*)d_in[3];
    const float* Wkv   = (const float*)d_in[4];
    const float* bkv   = (const float*)d_in[5];
    const float* Wg    = (const float*)d_in[6];
    const float* bg    = (const float*)d_in[7];
    const float* vinit = (const float*)d_in[8];
    const float* Wm    = (const float*)d_in[9];
    const float* bm    = (const float*)d_in[10];

    char* ws = (char*)d_ws;
    size_t off = 0;
    auto alloc = [&](size_t bytes) {
        void* p = ws + off;
        off += (bytes + 255) & ~(size_t)255;
        return p;
    };
    __bf16* WqT  = (__bf16*)alloc((size_t)512 * 512 * 2);
    __bf16* WkvT = (__bf16*)alloc((size_t)1024 * 512 * 2);
    __bf16* WgT  = (__bf16*)alloc((size_t)512 * 512 * 2);
    __bf16* WmT  = (__bf16*)alloc((size_t)512 * 512 * 2);
    float*  tvv  = (float*)alloc((size_t)24 * 512 * 4);
    const size_t MC = (size_t)49152 * 512;
    __bf16* qhb  = (__bf16*)alloc(MC * 2);
    __bf16* khb  = (__bf16*)alloc(MC * 2);
    __bf16* vlin = (__bf16*)alloc(MC * 2);
    __bf16* vsb  = (__bf16*)alloc(MC * 2);
    __bf16* xsb  = vlin;  // v_lin fully consumed before attention writes x

    const dim3 tb(32, 8);
    k_transpose<<<dim3(16, 16), tb, 0, stream>>>(Wq,  WqT,  512, 512);
    k_transpose<<<dim3(32, 16), tb, 0, stream>>>(Wkv, WkvT, 512, 1024);
    k_transpose<<<dim3(16, 16), tb, 0, stream>>>(Wg,  WgT,  512, 512);
    k_transpose<<<dim3(16, 16), tb, 0, stream>>>(Wm,  WmT,  512, 512);
    k_tanhv<<<48, 256, 0, stream>>>(vinit, tvv, 24 * 512);

    // qh = l2norm_per_head(q@Wq + bq)
    k_gemm<0, true ><<<dim3(384, 8),  256, 0, stream>>>(q,    WqT,  bq,  qhb,   nullptr, nullptr);
    // kh = l2norm_per_head(k_lin), v_lin passthrough
    k_gemm<1, true ><<<dim3(384, 16), 256, 0, stream>>>(kv,   WkvT, bkv, khb,   vlin,    nullptr);
    // v = l2norm_per_head(relu(sigmoid(v_lin@Wg + bg) * tanh(v_init)))
    k_gemm<2, false><<<dim3(384, 8),  256, 0, stream>>>(vlin, WgT,  bg,  vsb,   nullptr, tvv);
    // x = softmax(q k^T / 8) v  per (bt, h)
    k_attn<<<8192, 64, 0, stream>>>(qhb, khb, vsb, xsb);
    // out = x@Wm + bm + shortcut(q)
    k_gemm<3, false><<<dim3(384, 8),  256, 0, stream>>>(xsb,  WmT,  bm,  d_out, nullptr, q);
}
